// SimCLR_Loss_45792941310092
// MI455X (gfx1250) — compile-verified, matrix operand order checked by reading, not run on hardware
//
#include <hip/hip_runtime.h>
#include <hip/hip_bf16.h>
#include <math.h>

typedef __attribute__((ext_vector_type(16))) _Float16 v16h;
typedef __attribute__((ext_vector_type(8)))  _Float16 v8h;
typedef __attribute__((ext_vector_type(8)))  float    v8f;

#define BATCH   128
#define TSTEPS  256
#define FDIM    512
#define NROWS   256      // 2*BATCH
#define KCHUNK  128
#define NCHUNK  (FDIM / KCHUNK)
#define PITCH   136      // halves per LDS row (272B pitch -> 4-bank shift/row, conflict-free)
#define NWAVE   16
#define NTHR    512

// LDS: raw f16 tile [256][PITCH] + 256 f32 reciprocal norms.
// Epilogue partial-LSE arrays alias the (dead) tile.
#define SMEM_TILE_BYTES (NROWS * PITCH * 2)
#define SMEM_BYTES      (SMEM_TILE_BYTES + NROWS * 4)

__device__ __forceinline__ const float* row_base(const float* zi, const float* zj,
                                                 int row, int t) {
    return (row < BATCH)
        ? (zi + (size_t)row * TSTEPS * FDIM + (size_t)t * FDIM)
        : (zj + (size_t)(row - BATCH) * TSTEPS * FDIM + (size_t)t * FDIM);
}

__global__ __launch_bounds__(NTHR)
void simclr_sim_kernel(const float* __restrict__ zi, const float* __restrict__ zj,
                       float* __restrict__ per_t) {
    extern __shared__ char smem[];
    _Float16* tile   = (_Float16*)smem;
    float*    rscale = (float*)(smem + SMEM_TILE_BYTES);

    const int t       = blockIdx.x;
    const int tid     = threadIdx.x;
    const int lane    = tid & 31;
    const int w       = tid >> 5;        // wave 0..15
    const int rg      = w >> 2;          // row group: rows [64*rg, 64*rg+64)
    const int cg      = w & 3;           // col group: cols [64*cg, 64*cg+64)
    const int lo      = lane & 15;
    const int halfoff = (lane >= 16) ? 8 : 0;

    // Fill-pass identity: each thread owns half of one row for all chunks.
    const int frow = tid >> 1;
    const int fseg = (tid & 1) * 64;
    const float* frp = row_base(zi, zj, frow, t) + fseg;

    v8f   acc[4][4] = {};
    float ss = 0.f;                      // raw row-half sum of squares

    for (int c = 0; c < NCHUNK; ++c) {
        // ---- cooperative raw f16 fill + fused sum-of-squares -------------
        {
            const float4* rp4 = (const float4*)(frp + c * KCHUNK);
            _Float16* dst = tile + frow * PITCH + fseg;
            #pragma unroll
            for (int j = 0; j < 8; ++j) {
                float4 a = rp4[2 * j + 0];
                float4 b = rp4[2 * j + 1];
                ss += a.x * a.x + a.y * a.y + a.z * a.z + a.w * a.w;
                ss += b.x * b.x + b.y * b.y + b.z * b.z + b.w * b.w;
                v8h h;
                h[0] = (_Float16)a.x; h[1] = (_Float16)a.y;
                h[2] = (_Float16)a.z; h[3] = (_Float16)a.w;
                h[4] = (_Float16)b.x; h[5] = (_Float16)b.y;
                h[6] = (_Float16)b.z; h[7] = (_Float16)b.w;
                *(v8h*)(dst + 8 * j) = h;
            }
            if (c + 1 < NCHUNK)                      // warm caches for next chunk
                __builtin_prefetch(rp4 + 32, 0, 0);  // -> global_prefetch_b8
            if (c == NCHUNK - 1) {
                // combine the two half-row partials (adjacent lanes) -> 1/||z||
                float sstot = ss + __shfl_xor(ss, 1, 32);
                if ((tid & 1) == 0)
                    rscale[frow] = 1.0f / fmaxf(sqrtf(sstot), 1e-8f);
            }
        }
        __syncthreads();

        // ---- 4x4-tile WMMA over this K-chunk -----------------------------
        #pragma unroll
        for (int kt = 0; kt < KCHUNK / 32; ++kt) {
            // per-lane K offset per the 16-bit A 16x32 VGPR layout
            const int kb = kt * 32 + ((lane < 16) ? 0 : 8);
            v16h bf[4];
            #pragma unroll
            for (int j = 0; j < 4; ++j) {
                // B = Z^T (col-major) -> same gather pattern, row = column index
                const _Float16* bp = tile + (cg * 64 + j * 16 + lo) * PITCH + kb;
                *((v8h*)&bf[j])       = *(const v8h*)(bp);
                *(((v8h*)&bf[j]) + 1) = *(const v8h*)(bp + 16);
            }
            v16h a_cur;
            {
                const _Float16* ap = tile + (rg * 64 + lo) * PITCH + kb;
                *((v8h*)&a_cur)       = *(const v8h*)(ap);
                *(((v8h*)&a_cur) + 1) = *(const v8h*)(ap + 16);
            }
            #pragma unroll
            for (int i = 0; i < 4; ++i) {
                v16h a_nxt;
                if (i < 3) {   // software-pipeline next A frag over the WMMAs
                    const _Float16* ap =
                        tile + (rg * 64 + (i + 1) * 16 + lo) * PITCH + kb;
                    *((v8h*)&a_nxt)       = *(const v8h*)(ap);
                    *(((v8h*)&a_nxt) + 1) = *(const v8h*)(ap + 16);
                }
                #pragma unroll
                for (int j = 0; j < 4; ++j)
                    acc[i][j] = __builtin_amdgcn_wmma_f32_16x16x32_f16(
                        false, a_cur, false, bf[j], (short)0, acc[i][j],
                        false, false);
                a_cur = a_nxt;
            }
        }
        __syncthreads();
    }

    // ---- Epilogue: post-scale by rs_i*rs_j/T, masked partial LSE ----------
    // C/D layout: VGPR r, lanes 0-15 -> local row r, lanes 16-31 -> r+8,
    // column = 16*j + lo (within this wave's col group).
    float* pmax = (float*)smem;          // [256][4] partial row max per colgroup
    float* psum = pmax + NROWS * 4;      // [256][4] partial sum(exp(v-mx))
    float* ppos = psum + NROWS * 4;      // [256]    positive-pair logit
    float* rl   = ppos + NROWS;          // [256]    per-row loss

    const float INV_T = 2.0f;            // 1 / temperature(0.5)
    float rsc[4];
    #pragma unroll
    for (int j = 0; j < 4; ++j) rsc[j] = rscale[cg * 64 + j * 16 + lo];

    #pragma unroll
    for (int i = 0; i < 4; ++i) {
        #pragma unroll
        for (int r = 0; r < 8; ++r) {
            const int   R    = rg * 64 + i * 16 + r + halfoff;
            const int   Rpos = R ^ 128;
            const float rowf = INV_T * rscale[R];    // broadcast LDS load
            float mx = -1e30f, pos = 0.f;
            bool haspos = false;
            float v[4];
            #pragma unroll
            for (int j = 0; j < 4; ++j) {
                const int col = cg * 64 + j * 16 + lo;
                v[j] = rowf * rsc[j] * acc[i][j][r];
                if (col != R)    mx = fmaxf(mx, v[j]);  // diagonal masked
                if (col == Rpos) { pos = v[j]; haspos = true; }
            }
            #pragma unroll
            for (int m = 1; m < 16; m <<= 1) mx = fmaxf(mx, __shfl_xor(mx, m, 32));
            float s = 0.f;
            #pragma unroll
            for (int j = 0; j < 4; ++j) {
                const int col = cg * 64 + j * 16 + lo;
                if (col != R) s += __expf(v[j] - mx);
            }
            #pragma unroll
            for (int m = 1; m < 16; m <<= 1) s += __shfl_xor(s, m, 32);
            if (lo == (r + halfoff)) {               // one lane per half-row
                pmax[R * 4 + cg] = mx;
                psum[R * 4 + cg] = s;
            }
            if (haspos) ppos[R] = pos;               // exactly one lane globally
        }
    }
    __syncthreads();

    if (tid < NROWS) {
        float m = pmax[tid * 4 + 0];
        #pragma unroll
        for (int g = 1; g < 4; ++g) m = fmaxf(m, pmax[tid * 4 + g]);
        float s = 0.f;
        #pragma unroll
        for (int g = 0; g < 4; ++g)
            s += psum[tid * 4 + g] * __expf(pmax[tid * 4 + g] - m);
        rl[tid] = __logf(s) + m - ppos[tid];
    }
    __syncthreads();
    for (int s2 = 128; s2 > 0; s2 >>= 1) {
        if (tid < s2) rl[tid] += rl[tid + s2];
        __syncthreads();
    }
    if (tid == 0) per_t[t] = rl[0];
}

__global__ __launch_bounds__(256)
void simclr_reduce_kernel(const float* __restrict__ per_t, float* __restrict__ out) {
    __shared__ float sm[256];
    sm[threadIdx.x] = per_t[threadIdx.x];
    __syncthreads();
    for (int s = 128; s > 0; s >>= 1) {
        if (threadIdx.x < s) sm[threadIdx.x] += sm[threadIdx.x + s];
        __syncthreads();
    }
    if (threadIdx.x == 0)
        out[0] = sm[0] / ((float)TSTEPS * (float)NROWS);  // mean over t, /N
}

extern "C" void kernel_launch(void* const* d_in, const int* in_sizes, int n_in,
                              void* d_out, int out_size, void* d_ws, size_t ws_size,
                              hipStream_t stream) {
    (void)in_sizes; (void)n_in; (void)out_size; (void)ws_size;
    const float* zi = (const float*)d_in[0];
    const float* zj = (const float*)d_in[1];
    float* per_t = (float*)d_ws;          // 256 floats of scratch
    float* out   = (float*)d_out;

    (void)hipFuncSetAttribute((const void*)simclr_sim_kernel,
                              hipFuncAttributeMaxDynamicSharedMemorySize,
                              SMEM_BYTES);

    simclr_sim_kernel<<<TSTEPS, NTHR, SMEM_BYTES, stream>>>(zi, zj, per_t);
    simclr_reduce_kernel<<<1, 256, 0, stream>>>(per_t, out);
}
